// CNNMamba_29368986370376
// MI455X (gfx1250) — compile-verified, matrix operand order checked by reading, not run on hardware
//
#include <hip/hip_runtime.h>
#include <hip/hip_bf16.h>
#include <math.h>

// ---------------- model dims ----------------
#define B_SZ     8
#define NMELS    96
#define T_SZ     1024
#define NDIMS    192
#define CCH      32
#define DMODEL   1536
#define NLAYERS  4
#define DSTATE   16
#define DCONV    4
#define DINNER   3072
#define DTRANK   96
#define NCLASSES 5
#define MT       (B_SZ * T_SZ)        // 8192 token rows

typedef __bf16 bf16;
typedef __attribute__((ext_vector_type(16))) __bf16 v16bf;
typedef __attribute__((ext_vector_type(8)))  __bf16 v8bf;
typedef __attribute__((ext_vector_type(8)))  float  v8f;

union V16 { v16bf v; v8bf h[2]; };

__device__ __forceinline__ float silu_f(float x) { return x / (1.0f + __expf(-x)); }

// ---------------- CNN front-end ----------------

// stacked input: s[b, c, t], c<96 -> x, c>=96 -> relu(causal diff)
__global__ void k_stack(const float* __restrict__ x, float* __restrict__ s) {
  int idx = blockIdx.x * blockDim.x + threadIdx.x;
  const int total = B_SZ * NDIMS * T_SZ;
  if (idx >= total) return;
  int t = idx % T_SZ;
  int c = (idx / T_SZ) % NDIMS;
  int b = idx / (T_SZ * NDIMS);
  float v;
  if (c < NMELS) {
    v = x[(b * NMELS + c) * T_SZ + t];
  } else {
    int cc = c - NMELS;
    v = (t == 0) ? 0.0f
                 : x[(b * NMELS + cc) * T_SZ + t] - x[(b * NMELS + cc) * T_SZ + t - 1];
    v = fmaxf(v, 0.0f);
  }
  s[idx] = v;
}

// conv1a: 1->32ch 3x3 SAME on (192,1024), + bias, silu
__global__ void k_conv1a(const float* __restrict__ s, const float* __restrict__ w,
                         const float* __restrict__ bias, float* __restrict__ ha) {
  int idx = blockIdx.x * blockDim.x + threadIdx.x;
  const int total = B_SZ * CCH * NDIMS * T_SZ;
  if (idx >= total) return;
  int j  = idx % T_SZ;
  int i  = (idx / T_SZ) % NDIMS;
  int co = (idx / (T_SZ * NDIMS)) % CCH;
  int b  = idx / (T_SZ * NDIMS * CCH);
  float acc = bias[co];
  #pragma unroll
  for (int di = 0; di < 3; ++di) {
    int ii = i + di - 1;
    if (ii < 0 || ii >= NDIMS) continue;
    #pragma unroll
    for (int dj = 0; dj < 3; ++dj) {
      int jj = j + dj - 1;
      if (jj < 0 || jj >= T_SZ) continue;
      acc += w[co * 9 + di * 3 + dj] * s[(b * NDIMS + ii) * T_SZ + jj];
    }
  }
  ha[idx] = silu_f(acc);
}

// conv1b(32->32) + 1x1 skip of stacked input + silu, then maxpool H/2
__global__ void k_conv1b_pool(const float* __restrict__ ha, const float* __restrict__ s,
                              const float* __restrict__ wb, const float* __restrict__ bb,
                              const float* __restrict__ ws1, const float* __restrict__ bs1,
                              float* __restrict__ p1) {
  int idx = blockIdx.x * blockDim.x + threadIdx.x;
  const int HO = NDIMS / 2;  // 96
  const int total = B_SZ * CCH * HO * T_SZ;
  if (idx >= total) return;
  int j  = idx % T_SZ;
  int ho = (idx / T_SZ) % HO;
  int co = (idx / (T_SZ * HO)) % CCH;
  int b  = idx / (T_SZ * HO * CCH);
  float best = -INFINITY;
  for (int r2 = 0; r2 < 2; ++r2) {
    int i = ho * 2 + r2;
    float acc = bb[co];
    for (int ci = 0; ci < CCH; ++ci) {
      const float* hp = ha + (size_t)(b * CCH + ci) * NDIMS * T_SZ;
      #pragma unroll
      for (int di = 0; di < 3; ++di) {
        int ii = i + di - 1;
        if (ii < 0 || ii >= NDIMS) continue;
        #pragma unroll
        for (int dj = 0; dj < 3; ++dj) {
          int jj = j + dj - 1;
          if (jj < 0 || jj >= T_SZ) continue;
          acc += wb[((co * CCH + ci) * 3 + di) * 3 + dj] * hp[ii * T_SZ + jj];
        }
      }
    }
    acc += ws1[co] * s[(b * NDIMS + i) * T_SZ + j] + bs1[co];
    best = fmaxf(best, silu_f(acc));
  }
  p1[idx] = best;
}

// conv2a: 32->32 3x3 SAME on (96,1024), + bias, silu
__global__ void k_conv2a(const float* __restrict__ p1, const float* __restrict__ w,
                         const float* __restrict__ bias, float* __restrict__ ha2) {
  int idx = blockIdx.x * blockDim.x + threadIdx.x;
  const int H = 96;
  const int total = B_SZ * CCH * H * T_SZ;
  if (idx >= total) return;
  int j  = idx % T_SZ;
  int i  = (idx / T_SZ) % H;
  int co = (idx / (T_SZ * H)) % CCH;
  int b  = idx / (T_SZ * H * CCH);
  float acc = bias[co];
  for (int ci = 0; ci < CCH; ++ci) {
    const float* pp = p1 + (size_t)(b * CCH + ci) * H * T_SZ;
    #pragma unroll
    for (int di = 0; di < 3; ++di) {
      int ii = i + di - 1;
      if (ii < 0 || ii >= H) continue;
      #pragma unroll
      for (int dj = 0; dj < 3; ++dj) {
        int jj = j + dj - 1;
        if (jj < 0 || jj >= T_SZ) continue;
        acc += w[((co * CCH + ci) * 3 + di) * 3 + dj] * pp[ii * T_SZ + jj];
      }
    }
  }
  ha2[idx] = silu_f(acc);
}

// conv2b + identity skip + silu + maxpool H/2, fused reshape+transpose store:
// h_res[(b*T + j)*DMODEL + co*48 + ho]
__global__ void k_conv2b_pool_tr(const float* __restrict__ ha2, const float* __restrict__ p1,
                                 const float* __restrict__ wb, const float* __restrict__ bb,
                                 float* __restrict__ h_res) {
  int idx = blockIdx.x * blockDim.x + threadIdx.x;
  const int H = 96, HO = 48;
  const int total = B_SZ * CCH * HO * T_SZ;
  if (idx >= total) return;
  int j  = idx % T_SZ;
  int ho = (idx / T_SZ) % HO;
  int co = (idx / (T_SZ * HO)) % CCH;
  int b  = idx / (T_SZ * HO * CCH);
  float best = -INFINITY;
  for (int r2 = 0; r2 < 2; ++r2) {
    int i = ho * 2 + r2;
    float acc = bb[co];
    for (int ci = 0; ci < CCH; ++ci) {
      const float* hp = ha2 + (size_t)(b * CCH + ci) * H * T_SZ;
      #pragma unroll
      for (int di = 0; di < 3; ++di) {
        int ii = i + di - 1;
        if (ii < 0 || ii >= H) continue;
        #pragma unroll
        for (int dj = 0; dj < 3; ++dj) {
          int jj = j + dj - 1;
          if (jj < 0 || jj >= T_SZ) continue;
          acc += wb[((co * CCH + ci) * 3 + di) * 3 + dj] * hp[ii * T_SZ + jj];
        }
      }
    }
    acc += p1[(size_t)(b * CCH + co) * H * T_SZ + i * T_SZ + j];  // identity skip
    best = fmaxf(best, silu_f(acc));
  }
  h_res[((size_t)(b * T_SZ) + j) * DMODEL + co * HO + ho] = best;
}

// ---------------- Mamba building blocks ----------------

__global__ void k_cvt_bf16(const float* __restrict__ in, bf16* __restrict__ out, int n) {
  int idx = blockIdx.x * blockDim.x + threadIdx.x;
  if (idx < n) out[idx] = (bf16)in[idx];
}

// rmsnorm over last dim (1536) -> bf16
__global__ void k_rmsnorm(const float* __restrict__ h, const float* __restrict__ w,
                          bf16* __restrict__ xn) {
  __shared__ float red[256];
  int row = blockIdx.x;
  const float* hr = h + (size_t)row * DMODEL;
  float ss = 0.0f;
  for (int d = threadIdx.x; d < DMODEL; d += 256) { float v = hr[d]; ss += v * v; }
  red[threadIdx.x] = ss;
  __syncthreads();
  for (int st = 128; st > 0; st >>= 1) {
    if ((int)threadIdx.x < st) red[threadIdx.x] += red[threadIdx.x + st];
    __syncthreads();
  }
  float scale = rsqrtf(red[0] / (float)DMODEL + 1e-5f);
  bf16* xr = xn + (size_t)row * DMODEL;
  for (int d = threadIdx.x; d < DMODEL; d += 256) xr[d] = (bf16)(hr[d] * scale * w[d]);
}

// WMMA bf16 GEMM: C[M,N] = A[M,K] (row-major) * W[N,K]^T (weights row-major over K).
// Block = 128 threads = 4 waves arranged 2x2 over a 64x128 block tile.
// Each wave owns a 32x64 macro-tile = 2x4 v_wmma_f32_16x16x32_bf16 accumulators:
// per K-step 6 b128-load-pairs feed 8 WMMAs (21 FLOP/B from cache vs 16 for 2x2).
// MODE 0: store bf16. MODE 1: store f32 with residual add. MODE 2: softplus(acc + bias[n]) f32.
template <int MODE>
__global__ void k_gemm(const bf16* __restrict__ A, int lda,
                       const bf16* __restrict__ W, int ldb,
                       void* Cout, int ldc,
                       const float* __restrict__ bias,
                       const float* resid,
                       int M, int N, int K) {
  const int lane = threadIdx.x & 31;
  const int wave = threadIdx.x >> 5;
  const int wm = wave >> 1;           // 0..1
  const int wn = wave & 1;            // 0..1
  const int m0 = blockIdx.y * 64 + wm * 32;
  const int n0 = blockIdx.x * 128 + wn * 64;
  if (m0 >= M || n0 >= N) return;     // wave-uniform
  const int l16  = lane & 15;
  const int koff = (lane >> 4) * 8;   // lanes 0-15: K 0..7/16..23; lanes 16-31: K 8..15/24..31

  const bf16* arow[2];
  arow[0] = A + (size_t)(m0 + l16) * lda;
  arow[1] = A + (size_t)(m0 + 16 + l16) * lda;
  const bf16* brow[4];
  #pragma unroll
  for (int jb = 0; jb < 4; ++jb) brow[jb] = W + (size_t)(n0 + jb * 16 + l16) * ldb;

  v8f acc[2][4];
  #pragma unroll
  for (int i = 0; i < 2; ++i)
    #pragma unroll
    for (int j = 0; j < 4; ++j) acc[i][j] = (v8f){};

  for (int k = 0; k < K; k += 32) {
    const int ka = k + koff;
    V16 a[2], b[4];
    #pragma unroll
    for (int i = 0; i < 2; ++i) {
      a[i].h[0] = *(const v8bf*)(arow[i] + ka);
      a[i].h[1] = *(const v8bf*)(arow[i] + ka + 16);
    }
    #pragma unroll
    for (int j = 0; j < 4; ++j) {
      b[j].h[0] = *(const v8bf*)(brow[j] + ka);
      b[j].h[1] = *(const v8bf*)(brow[j] + ka + 16);
    }
    #pragma unroll
    for (int i = 0; i < 2; ++i)
      #pragma unroll
      for (int j = 0; j < 4; ++j)
        acc[i][j] = __builtin_amdgcn_wmma_f32_16x16x32_bf16(
            false, a[i].v, false, b[j].v, (short)0, acc[i][j], false, false);
  }

  // C/D layout: lane<16 -> N=n_base+lane, VGPR v -> M=m_base+v; lane>=16 -> M=m_base+8+v
  const int nc = lane & 15;
  const int mc = (lane >> 4) * 8;
  #pragma unroll
  for (int i = 0; i < 2; ++i) {
    #pragma unroll
    for (int j = 0; j < 4; ++j) {
      const int mbase = m0 + i * 16 + mc;
      const int n = n0 + j * 16 + nc;
      #pragma unroll
      for (int v = 0; v < 8; ++v) {
        size_t idx = (size_t)(mbase + v) * ldc + n;
        float val = acc[i][j][v];
        if constexpr (MODE == 0) {
          ((bf16*)Cout)[idx] = (bf16)val;
        } else if constexpr (MODE == 1) {
          ((float*)Cout)[idx] = resid[idx] + val;
        } else {
          float xv = val + bias[n];
          ((float*)Cout)[idx] = (xv > 20.0f) ? xv : log1pf(__expf(xv));
        }
      }
    }
  }
}

// depthwise causal conv (width 4) + bias + silu, reading u-half of xz (bf16) -> u bf16
__global__ void k_dwconv(const bf16* __restrict__ xz, const float* __restrict__ cw,
                         const float* __restrict__ cb, bf16* __restrict__ u) {
  int idx = blockIdx.x * blockDim.x + threadIdx.x;
  const int total = MT * DINNER;
  if (idx >= total) return;
  int d = idx % DINNER;
  int bt = idx / DINNER;
  int t = bt % T_SZ;
  int b = bt / T_SZ;
  float acc = cb[d];
  #pragma unroll
  for (int k = 0; k < DCONV; ++k) {
    int tt = t - (DCONV - 1) + k;
    if (tt >= 0)
      acc += cw[d * DCONV + k] * (float)xz[((size_t)(b * T_SZ + tt)) * (2 * DINNER) + d];
  }
  u[(size_t)bt * DINNER + d] = (bf16)silu_f(acc);
}

// selective scan: one thread per (b,d); h[16] in registers, 1024 sequential steps.
// Latency-bound: prefetch next timestep's rows (global_prefetch_b8) to overlap the
// dependent-load latency with the 16 v_exp_f32 TRANS ops of the current step.
// fused: y = scan + u*Dp; y *= silu(z); store bf16 for out-proj GEMM.
__global__ void k_scan(const float* __restrict__ delta, const bf16* __restrict__ u,
                       const bf16* __restrict__ xdbl,   // (MT,128): [0:96]=dt, [96:112]=B, [112:128]=C
                       const bf16* __restrict__ xz,     // (MT,6144): z at cols 3072..6143
                       const float* __restrict__ A_log, const float* __restrict__ Dp,
                       bf16* __restrict__ ybf) {
  int tid = blockIdx.x * blockDim.x + threadIdx.x;
  if (tid >= B_SZ * DINNER) return;
  int d = tid % DINNER;
  int b = tid / DINNER;
  float Arow[DSTATE];
  #pragma unroll
  for (int s = 0; s < DSTATE; ++s) Arow[s] = -__expf(A_log[(size_t)d * DSTATE + s]);
  const float Dv = Dp[d];
  float h[DSTATE];
  #pragma unroll
  for (int s = 0; s < DSTATE; ++s) h[s] = 0.0f;

  for (int t = 0; t < T_SZ; ++t) {
    size_t row = (size_t)(b * T_SZ + t);
    if (t + 1 < T_SZ) {
      size_t nrow = row + 1;
      __builtin_prefetch(delta + nrow * DINNER + d, 0, 1);
      __builtin_prefetch(u + nrow * DINNER + d, 0, 1);
      __builtin_prefetch(xdbl + nrow * 128 + DTRANK, 0, 1);
    }
    float dlt = delta[row * DINNER + d];
    float uu  = (float)u[row * DINNER + d];
    float du  = dlt * uu;
    const bf16* bc = xdbl + row * 128 + DTRANK;
    float y = 0.0f;
    #pragma unroll
    for (int s = 0; s < DSTATE; ++s) {
      float dA = __expf(dlt * Arow[s]);
      h[s] = dA * h[s] + du * (float)bc[s];
      y += h[s] * (float)bc[DSTATE + s];
    }
    float yy = y + uu * Dv;
    float zz = (float)xz[row * (2 * DINNER) + DINNER + d];
    ybf[row * DINNER + d] = (bf16)(yy * silu_f(zz));
  }
}

// final FC (1536 -> 5) + bias + silu, transposed store out[b, c, t]
__global__ void k_fc(const float* __restrict__ h, const float* __restrict__ fw,
                     const float* __restrict__ fb, float* __restrict__ out) {
  int idx = blockIdx.x * blockDim.x + threadIdx.x;
  const int total = B_SZ * NCLASSES * T_SZ;
  if (idx >= total) return;
  int t = idx % T_SZ;
  int c = (idx / T_SZ) % NCLASSES;
  int b = idx / (T_SZ * NCLASSES);
  const float* hr = h + ((size_t)(b * T_SZ) + t) * DMODEL;
  const float* wr = fw + c * DMODEL;
  float acc = fb[c];
  for (int d = 0; d < DMODEL; ++d) acc += hr[d] * wr[d];
  out[idx] = silu_f(acc);
}

// ---------------- workspace layout (bytes) ----------------
#define OFF_H    ((size_t)0)                         // 50331648 : h_res f32 (s reuses first 6.3MB)
#define OFF_BIG  ((size_t)50331648)                  // 201326592: ha | (xz_bf16 + u_bf16 + y_bf16)
#define OFF_P1   ((size_t)251658240)                 // 100663296: p1 f32 | delta f32
#define OFF_AUX  ((size_t)352321536)                 // 100663296: ha2 | (xn + bf16 weights + xdbl)
// within BIG (mamba phase)
#define OFFB_XZ   ((size_t)0)
#define OFFB_U    ((size_t)100663296)
#define OFFB_Y    ((size_t)150994944)
// within AUX (mamba phase)
#define OFFA_XN   ((size_t)0)
#define OFFA_WIN  ((size_t)25165824)
#define OFFA_WX   ((size_t)44040192)
#define OFFA_WDT  ((size_t)44826624)
#define OFFA_WOUT ((size_t)45416448)
#define OFFA_XDBL ((size_t)54853632)

extern "C" void kernel_launch(void* const* d_in, const int* in_sizes, int n_in,
                              void* d_out, int out_size, void* d_ws, size_t ws_size,
                              hipStream_t stream) {
  const float* x       = (const float*)d_in[0];
  const float* c1a_w   = (const float*)d_in[1];
  const float* c1a_b   = (const float*)d_in[2];
  const float* c1b_w   = (const float*)d_in[3];
  const float* c1b_b   = (const float*)d_in[4];
  const float* c1s_w   = (const float*)d_in[5];
  const float* c1s_b   = (const float*)d_in[6];
  const float* c2a_w   = (const float*)d_in[7];
  const float* c2a_b   = (const float*)d_in[8];
  const float* c2b_w   = (const float*)d_in[9];
  const float* c2b_b   = (const float*)d_in[10];
  const float* norm_w  = (const float*)d_in[11];
  const float* in_w    = (const float*)d_in[12];
  const float* conv_w  = (const float*)d_in[13];
  const float* conv_b  = (const float*)d_in[14];
  const float* xproj_w = (const float*)d_in[15];
  const float* dt_w    = (const float*)d_in[16];
  const float* dt_b    = (const float*)d_in[17];
  const float* A_log   = (const float*)d_in[18];
  const float* Dp      = (const float*)d_in[19];
  const float* out_w   = (const float*)d_in[20];
  const float* fc_w    = (const float*)d_in[21];
  const float* fc_b    = (const float*)d_in[22];
  float* out = (float*)d_out;

  char* ws = (char*)d_ws;
  float* h_res = (float*)(ws + OFF_H);
  float* s     = (float*)(ws + OFF_H);      // stacked input (dead before h_res is written)
  float* ha    = (float*)(ws + OFF_BIG);
  float* p1    = (float*)(ws + OFF_P1);
  float* ha2   = (float*)(ws + OFF_AUX);
  float* delta = (float*)(ws + OFF_P1);
  bf16* xz_bf   = (bf16*)(ws + OFF_BIG + OFFB_XZ);
  bf16* u_bf    = (bf16*)(ws + OFF_BIG + OFFB_U);
  bf16* y_bf    = (bf16*)(ws + OFF_BIG + OFFB_Y);
  bf16* xn_bf   = (bf16*)(ws + OFF_AUX + OFFA_XN);
  bf16* w_in    = (bf16*)(ws + OFF_AUX + OFFA_WIN);
  bf16* w_x     = (bf16*)(ws + OFF_AUX + OFFA_WX);
  bf16* w_dt    = (bf16*)(ws + OFF_AUX + OFFA_WDT);
  bf16* w_out   = (bf16*)(ws + OFF_AUX + OFFA_WOUT);
  bf16* xdbl_bf = (bf16*)(ws + OFF_AUX + OFFA_XDBL);

  const int TB = 256;
  auto blocks = [](int n, int tb) { return (n + tb - 1) / tb; };

  // ---- CNN front-end ----
  k_stack<<<blocks(B_SZ * NDIMS * T_SZ, TB), TB, 0, stream>>>(x, s);
  k_conv1a<<<blocks(B_SZ * CCH * NDIMS * T_SZ, TB), TB, 0, stream>>>(s, c1a_w, c1a_b, ha);
  k_conv1b_pool<<<blocks(B_SZ * CCH * 96 * T_SZ, TB), TB, 0, stream>>>(ha, s, c1b_w, c1b_b, c1s_w, c1s_b, p1);
  k_conv2a<<<blocks(B_SZ * CCH * 96 * T_SZ, TB), TB, 0, stream>>>(p1, c2a_w, c2a_b, ha2);
  k_conv2b_pool_tr<<<blocks(B_SZ * CCH * 48 * T_SZ, TB), TB, 0, stream>>>(ha2, p1, c2b_w, c2b_b, h_res);

  // ---- Mamba layers ----
  for (int l = 0; l < NLAYERS; ++l) {
    const float* norm_w_l  = norm_w + (size_t)l * DMODEL;
    const float* in_w_l    = in_w + (size_t)l * 2 * DINNER * DMODEL;
    const float* conv_w_l  = conv_w + (size_t)l * DINNER * DCONV;
    const float* conv_b_l  = conv_b + (size_t)l * DINNER;
    const float* xproj_w_l = xproj_w + (size_t)l * (DTRANK + 2 * DSTATE) * DINNER;
    const float* dt_w_l    = dt_w + (size_t)l * DINNER * DTRANK;
    const float* dt_b_l    = dt_b + (size_t)l * DINNER;
    const float* A_log_l   = A_log + (size_t)l * DINNER * DSTATE;
    const float* Dp_l      = Dp + (size_t)l * DINNER;
    const float* out_w_l   = out_w + (size_t)l * DMODEL * DINNER;

    k_cvt_bf16<<<blocks(2 * DINNER * DMODEL, TB), TB, 0, stream>>>(in_w_l, w_in, 2 * DINNER * DMODEL);
    k_cvt_bf16<<<blocks((DTRANK + 2 * DSTATE) * DINNER, TB), TB, 0, stream>>>(xproj_w_l, w_x, (DTRANK + 2 * DSTATE) * DINNER);
    k_cvt_bf16<<<blocks(DINNER * DTRANK, TB), TB, 0, stream>>>(dt_w_l, w_dt, DINNER * DTRANK);
    k_cvt_bf16<<<blocks(DMODEL * DINNER, TB), TB, 0, stream>>>(out_w_l, w_out, DMODEL * DINNER);

    k_rmsnorm<<<MT, 256, 0, stream>>>(h_res, norm_w_l, xn_bf);

    // in-proj: (8192 x 1536) * (6144 x 1536)^T -> xz bf16
    {
      dim3 g(2 * DINNER / 128, MT / 64);
      k_gemm<0><<<g, 128, 0, stream>>>(xn_bf, DMODEL, w_in, DMODEL,
                                       (void*)xz_bf, 2 * DINNER, nullptr, nullptr,
                                       MT, 2 * DINNER, DMODEL);
    }

    k_dwconv<<<blocks(MT * DINNER, TB), TB, 0, stream>>>(xz_bf, conv_w_l, conv_b_l, u_bf);

    // x-proj: (8192 x 3072) * (128 x 3072)^T -> xdbl bf16
    {
      dim3 g(1, MT / 64);
      k_gemm<0><<<g, 128, 0, stream>>>(u_bf, DINNER, w_x, DINNER,
                                       (void*)xdbl_bf, 128, nullptr, nullptr,
                                       MT, 128, DINNER);
    }

    // dt-proj: (8192 x 96) * (3072 x 96)^T + dt_b -> softplus -> delta f32
    {
      dim3 g(DINNER / 128, MT / 64);
      k_gemm<2><<<g, 128, 0, stream>>>(xdbl_bf, 128, w_dt, DTRANK,
                                       (void*)delta, DINNER, dt_b_l, nullptr,
                                       MT, DINNER, DTRANK);
    }

    k_scan<<<blocks(B_SZ * DINNER, TB), TB, 0, stream>>>(delta, u_bf, xdbl_bf, xz_bf,
                                                         A_log_l, Dp_l, y_bf);

    // out-proj: (8192 x 3072) * (1536 x 3072)^T + residual -> h_res f32
    {
      dim3 g(DMODEL / 128, MT / 64);
      k_gemm<1><<<g, 128, 0, stream>>>(y_bf, DINNER, w_out, DINNER,
                                       (void*)h_res, DMODEL, nullptr, h_res,
                                       MT, DMODEL, DINNER);
    }
  }

  // ---- classifier head ----
  k_fc<<<blocks(B_SZ * NCLASSES * T_SZ, TB), TB, 0, stream>>>(h_res, fc_w, fc_b, out);

  (void)in_sizes; (void)n_in; (void)out_size; (void)ws_size;
}
